// FeatureAggregationModule_335007449988
// MI455X (gfx1250) — matrix operand order repaired and checked by reading, not compiled, not run on hardware
//
#include <hip/hip_runtime.h>

// ---------------- problem constants ----------------
#define B_    16
#define CIN_  256
#define COUT_ 256
#define CQ_   32
#define N_    9216     // 96*96
#define EPS_ATT 1e-6f
#define EPS_BN  1e-5f

typedef __attribute__((ext_vector_type(16))) __bf16 bf16x16;
typedef __attribute__((ext_vector_type(8)))  float  f32x8;

union Frag {
  uint4   u[2];   // two 16-byte chunks
  bf16x16 v;      // 16 bf16 = 8 VGPRs
};
union Pack8 { __bf16 h[8]; uint4 u; };   // 16B of bf16
union Pack4 { __bf16 h[4]; uint2 u; };   // 8B of bf16

__device__ __forceinline__ f32x8 wmma_bf16(const Frag& a, const Frag& b, f32x8 c) {
  // v_wmma_f32_16x16x32_bf16: (neg_a, A, neg_b, B, c_mod, C, reuse_a, reuse_b)
  return __builtin_amdgcn_wmma_f32_16x16x32_bf16(false, a.v, false, b.v, (short)0, c, false, false);
}

// async copy 16B global -> LDS (ASYNCcnt-tracked, no VGPR staging)
__device__ __forceinline__ void async_g2l_b128(const void* g, void* lds_generic) {
  // generic pointer to LDS: low 32 bits are the LDS byte offset (shared aperture)
  unsigned lo = (unsigned)(uintptr_t)lds_generic;
  asm volatile("global_load_async_to_lds_b128 %0, %1, off"
               :: "v"(lo), "v"(g) : "memory");
}
__device__ __forceinline__ void wait_asynccnt0() {
  asm volatile("s_wait_asynccnt 0" ::: "memory");
}

// =====================================================================
// Kernel 1: feat = relu(BN(conv1x1(concat(fsp,fcp))))
// GEMM per batch: D[256, N] = W[256,256] x X[256,N]
// block = 256 thr (8 waves). Tile BM=128, BN=128, BK=32, double-buffered LDS.
// Pipeline per K-step: barrier -> issue next-tile global loads (regs) ->
//   frag ds_loads + 8x WMMA -> cvt+ds_store next tile (loadcnt wait lands
//   after the WMMAs, hiding HBM latency behind the matrix pipe).
// Outputs: featF[b][c][n] (f32) and featT[b][n][c] (bf16, 16B-packed).
// =====================================================================
__global__ __launch_bounds__(256)
void k1_conv_bn_relu(const float* __restrict__ fsp, const float* __restrict__ fcp,
                     const float* __restrict__ convw,
                     const float* __restrict__ bns, const float* __restrict__ bnb,
                     const float* __restrict__ bnm, const float* __restrict__ bnv,
                     float* __restrict__ featF, __bf16* __restrict__ featT)
{
  const int n0 = blockIdx.x * 128;
  const int m0 = blockIdx.y * 128;
  const int b  = blockIdx.z;

  __shared__ __bf16 As[2][128 * 32];   // [m][k], row stride 32
  __shared__ __bf16 Bs[2][128 * 32];   // [n][k], row stride 32

  const int tid  = threadIdx.x;
  const int lane = tid & 31;
  const int wave = tid >> 5;
  const int wm   = (wave & 3) * 32;   // 4 waves along M
  const int wn   = (wave >> 2) * 64;  // 2 waves along N

  f32x8 acc[2][4];
#pragma unroll
  for (int i = 0; i < 2; ++i)
#pragma unroll
    for (int j = 0; j < 4; ++j) acc[i][j] = (f32x8){0,0,0,0,0,0,0,0};

  const int row  = lane & 15;
  const int kA   = (lane < 16) ? 0 : 8;
  const int kB   = (lane < 16) ? 0 : 8;

  // ---- prologue: stage K-tile 0 into buffer 0 ----
#pragma unroll
  for (int t = 0; t < 4; ++t) {
    int i  = tid + t * 256;
    int mi = i >> 3, kq = (i & 7) * 4;
    float4 w = *(const float4*)&convw[(size_t)(m0 + mi) * CIN_ + kq];
    Pack4 p; p.h[0] = (__bf16)w.x; p.h[1] = (__bf16)w.y;
             p.h[2] = (__bf16)w.z; p.h[3] = (__bf16)w.w;
    *(uint2*)&As[0][mi * 32 + kq] = p.u;
    int ki = i >> 5, nq = (i & 31) * 4;
    const float* src = (ki < 128) ? fsp : fcp;   // k0 == 0 here
    float4 v = *(const float4*)&src[((size_t)(b * 128 + (ki & 127))) * N_ + n0 + nq];
    Bs[0][(nq + 0) * 32 + ki] = (__bf16)v.x;
    Bs[0][(nq + 1) * 32 + ki] = (__bf16)v.y;
    Bs[0][(nq + 2) * 32 + ki] = (__bf16)v.z;
    Bs[0][(nq + 3) * 32 + ki] = (__bf16)v.w;
  }

  for (int kk = 0; kk < 8; ++kk) {
    const int cur = kk & 1, nxt = cur ^ 1;
    __syncthreads();   // tile kk visible; prior reads of buffer `nxt` done

    // issue next-tile global loads into registers (overlap with WMMAs)
    float4 aw[4], bx[4];
    if (kk < 7) {
      const int k0 = (kk + 1) * 32;
#pragma unroll
      for (int t = 0; t < 4; ++t) {
        int i  = tid + t * 256;
        int mi = i >> 3, kq = (i & 7) * 4;
        aw[t] = *(const float4*)&convw[(size_t)(m0 + mi) * CIN_ + k0 + kq];
        int ki = i >> 5, nq = (i & 31) * 4;
        int c  = k0 + ki;
        const float* src = (c < 128) ? fsp : fcp;
        bx[t] = *(const float4*)&src[((size_t)(b * 128 + (c & 127))) * N_ + n0 + nq];
      }
    }

    Frag a[2], bf[4];
#pragma unroll
    for (int i = 0; i < 2; ++i) {
      const __bf16* p = &As[cur][(wm + i * 16 + row) * 32];
      a[i].u[0] = *(const uint4*)(p + kA);
      a[i].u[1] = *(const uint4*)(p + kA + 16);
    }
#pragma unroll
    for (int j = 0; j < 4; ++j) {
      const __bf16* p = &Bs[cur][(wn + j * 16 + row) * 32];
      bf[j].u[0] = *(const uint4*)(p + kB);
      bf[j].u[1] = *(const uint4*)(p + kB + 16);
    }
#pragma unroll
    for (int i = 0; i < 2; ++i)
#pragma unroll
      for (int j = 0; j < 4; ++j)
        acc[i][j] = wmma_bf16(a[i], bf[j], acc[i][j]);

    // convert + store staged tile (loadcnt wait sits here, after the WMMAs)
    if (kk < 7) {
#pragma unroll
      for (int t = 0; t < 4; ++t) {
        int i  = tid + t * 256;
        int mi = i >> 3, kq = (i & 7) * 4;
        Pack4 p; p.h[0] = (__bf16)aw[t].x; p.h[1] = (__bf16)aw[t].y;
                 p.h[2] = (__bf16)aw[t].z; p.h[3] = (__bf16)aw[t].w;
        *(uint2*)&As[nxt][mi * 32 + kq] = p.u;
        int ki = i >> 5, nq = (i & 31) * 4;
        Bs[nxt][(nq + 0) * 32 + ki] = (__bf16)bx[t].x;
        Bs[nxt][(nq + 1) * 32 + ki] = (__bf16)bx[t].y;
        Bs[nxt][(nq + 2) * 32 + ki] = (__bf16)bx[t].z;
        Bs[nxt][(nq + 3) * 32 + ki] = (__bf16)bx[t].w;
      }
    }
  }

  // epilogue: BN (eval) + ReLU; featF [c][n] f32 + featT [n][c] bf16 packed
  const int cn   = lane & 15;
  const int roff = (lane < 16) ? 0 : 8;
#pragma unroll
  for (int i = 0; i < 2; ++i) {
    const int obase = m0 + wm + i * 16 + roff;
    float sc8[8], bi8[8];
#pragma unroll
    for (int r = 0; r < 8; ++r) {
      int o = obase + r;
      sc8[r] = bns[o] * rsqrtf(bnv[o] + EPS_BN);
      bi8[r] = bnb[o] - bnm[o] * sc8[r];
    }
#pragma unroll
    for (int j = 0; j < 4; ++j) {
      const int n = n0 + wn + j * 16 + cn;
      Pack8 p;
#pragma unroll
      for (int r = 0; r < 8; ++r) {
        float v = fmaxf(acc[i][j][r] * sc8[r] + bi8[r], 0.0f);
        featF[((size_t)(b * COUT_ + obase + r)) * N_ + n] = v;
        p.h[r] = (__bf16)v;
      }
      *(uint4*)&featT[((size_t)(b * N_ + n)) * COUT_ + obase] = p.u;
    }
  }
}

// =====================================================================
// Kernel 2: stacked QKV GEMM: rows 0..31 = Q, 32..63 = K, 64..319 = V
// D[320(pad 384), N] = Wqkv x feat[256, N]; Q,K -> f32, V -> bf16
// Double-buffered pipeline: B tile via GLOBAL_LOAD_ASYNC_TO_LDS_B128
// (issued after the barrier, waited at the top of the NEXT iteration ->
// DMA for tile k+1 overlaps the 8 WMMAs of tile k); A tile reg-staged.
// =====================================================================
__global__ __launch_bounds__(256)
void k2_qkv(const __bf16* __restrict__ featT,
            const float* __restrict__ qw, const float* __restrict__ qb,
            const float* __restrict__ kw, const float* __restrict__ kb,
            const float* __restrict__ vw, const float* __restrict__ vb,
            float* __restrict__ qF, float* __restrict__ kF, __bf16* __restrict__ vH)
{
  const int n0 = blockIdx.x * 128;
  const int m0 = blockIdx.y * 128;   // 0,128,256 (rows >=320 are padding)
  const int b  = blockIdx.z;

  __shared__ __bf16 As[2][128 * 32];
  __shared__ __bf16 Bs[2][128 * 32];    // [n][k], k contiguous -> 16B async chunks

  const int tid  = threadIdx.x;
  const int lane = tid & 31;
  const int wave = tid >> 5;
  const int wm   = (wave & 3) * 32;
  const int wn   = (wave >> 2) * 64;

  f32x8 acc[2][4];
#pragma unroll
  for (int i = 0; i < 2; ++i)
#pragma unroll
    for (int j = 0; j < 4; ++j) acc[i][j] = (f32x8){0,0,0,0,0,0,0,0};

  const int row  = lane & 15;
  const int kA   = (lane < 16) ? 0 : 8;
  const int kB   = (lane < 16) ? 0 : 8;

  // ---- prologue: stage K-tile 0 into buffer 0 ----
#pragma unroll
  for (int ch = tid; ch < 512; ch += 256) {
    int ni = ch >> 2;
    int kc = (ch & 3) * 8;
    const __bf16* g = featT + ((size_t)(b * N_ + n0 + ni)) * COUT_ + kc;
    async_g2l_b128(g, &Bs[0][ni * 32 + kc]);
  }
#pragma unroll
  for (int t = 0; t < 4; ++t) {
    int i  = tid + t * 256;
    int mi = i >> 3, kq = (i & 7) * 4;
    int o  = m0 + mi;
    float4 w;
    if      (o < 32)  w = *(const float4*)&qw[(size_t)o * COUT_ + kq];
    else if (o < 64)  w = *(const float4*)&kw[(size_t)(o - 32) * COUT_ + kq];
    else if (o < 320) w = *(const float4*)&vw[(size_t)(o - 64) * COUT_ + kq];
    else              w = make_float4(0.f, 0.f, 0.f, 0.f);
    Pack4 p; p.h[0] = (__bf16)w.x; p.h[1] = (__bf16)w.y;
             p.h[2] = (__bf16)w.z; p.h[3] = (__bf16)w.w;
    *(uint2*)&As[0][mi * 32 + kq] = p.u;
  }

  for (int kk = 0; kk < 8; ++kk) {
    const int cur = kk & 1, nxt = cur ^ 1;
    wait_asynccnt0();   // tile kk's DMA complete (tile kk+1 not yet issued)
    __syncthreads();    // all threads' tile-kk LDS writes visible

    float4 aw[4];
    if (kk < 7) {
      const int k0 = (kk + 1) * 32;
      // issue next-tile async DMA (runs while the WMMAs below execute)
#pragma unroll
      for (int ch = tid; ch < 512; ch += 256) {
        int ni = ch >> 2;
        int kc = (ch & 3) * 8;
        const __bf16* g = featT + ((size_t)(b * N_ + n0 + ni)) * COUT_ + k0 + kc;
        async_g2l_b128(g, &Bs[nxt][ni * 32 + kc]);
      }
#pragma unroll
      for (int t = 0; t < 4; ++t) {
        int i  = tid + t * 256;
        int mi = i >> 3, kq = (i & 7) * 4;
        int o  = m0 + mi;
        if      (o < 32)  aw[t] = *(const float4*)&qw[(size_t)o * COUT_ + k0 + kq];
        else if (o < 64)  aw[t] = *(const float4*)&kw[(size_t)(o - 32) * COUT_ + k0 + kq];
        else if (o < 320) aw[t] = *(const float4*)&vw[(size_t)(o - 64) * COUT_ + k0 + kq];
        else              aw[t] = make_float4(0.f, 0.f, 0.f, 0.f);
      }
    }

    Frag a[2], bf[4];
#pragma unroll
    for (int i = 0; i < 2; ++i) {
      const __bf16* p = &As[cur][(wm + i * 16 + row) * 32];
      a[i].u[0] = *(const uint4*)(p + kA);
      a[i].u[1] = *(const uint4*)(p + kA + 16);
    }
#pragma unroll
    for (int j = 0; j < 4; ++j) {
      const __bf16* p = &Bs[cur][(wn + j * 16 + row) * 32];
      bf[j].u[0] = *(const uint4*)(p + kB);
      bf[j].u[1] = *(const uint4*)(p + kB + 16);
    }
#pragma unroll
    for (int i = 0; i < 2; ++i)
#pragma unroll
      for (int j = 0; j < 4; ++j)
        acc[i][j] = wmma_bf16(a[i], bf[j], acc[i][j]);

    if (kk < 7) {
#pragma unroll
      for (int t = 0; t < 4; ++t) {
        int i  = tid + t * 256;
        int mi = i >> 3, kq = (i & 7) * 4;
        Pack4 p; p.h[0] = (__bf16)aw[t].x; p.h[1] = (__bf16)aw[t].y;
                 p.h[2] = (__bf16)aw[t].z; p.h[3] = (__bf16)aw[t].w;
        *(uint2*)&As[nxt][mi * 32 + kq] = p.u;
      }
    }
  }

  const int cn   = lane & 15;
  const int roff = (lane < 16) ? 0 : 8;
#pragma unroll
  for (int i = 0; i < 2; ++i) {
#pragma unroll
    for (int r = 0; r < 8; ++r) {
      int o = m0 + wm + i * 16 + roff + r;
#pragma unroll
      for (int j = 0; j < 4; ++j) {
        int n = n0 + wn + j * 16 + cn;
        float v = acc[i][j][r];
        if (o < 32) {
          qF[((size_t)(b * CQ_ + o)) * N_ + n] = v + qb[o];
        } else if (o < 64) {
          kF[((size_t)(b * CQ_ + (o - 32))) * N_ + n] = v + kb[o - 32];
        } else if (o < 320) {
          vH[((size_t)(b * COUT_ + (o - 64))) * N_ + n] = (__bf16)(v + vb[o - 64]);
        }
      }
    }
  }
}

// =====================================================================
// Kernel 3: per-column L2 norm of Q,K over 32 channels.
//   qnH[b][n][cq] (cq contiguous -> direct fragment loads), knH[b][cq][n],
//   ksum[b][cq] += sum_n Kn  (atomics; pre-zeroed)
// =====================================================================
__global__ __launch_bounds__(256)
void k3_norm_qk(const float* __restrict__ qF, const float* __restrict__ kF,
                __bf16* __restrict__ qnH, __bf16* __restrict__ knH,
                float* __restrict__ ksum)
{
  const int b = blockIdx.y;
  const int n = blockIdx.x * 256 + threadIdx.x;
  const int tid = threadIdx.x;

  float qv[CQ_], kv[CQ_], knv[CQ_];
  float qs = 0.f, ks = 0.f;
#pragma unroll
  for (int c = 0; c < CQ_; ++c) {
    qv[c] = qF[((size_t)(b * CQ_ + c)) * N_ + n];
    kv[c] = kF[((size_t)(b * CQ_ + c)) * N_ + n];
    qs += qv[c] * qv[c];
    ks += kv[c] * kv[c];
  }
  float qi = rsqrtf(qs);
  float ki = rsqrtf(ks);
  Pack8 pq[4];
#pragma unroll
  for (int c = 0; c < CQ_; ++c) {
    pq[c >> 3].h[c & 7] = (__bf16)(qv[c] * qi);
    knv[c] = kv[c] * ki;
    knH[((size_t)(b * CQ_ + c)) * N_ + n] = (__bf16)knv[c];
  }
#pragma unroll
  for (int q4 = 0; q4 < 4; ++q4)
    *(uint4*)&qnH[((size_t)(b * N_ + n)) * CQ_ + q4 * 8] = pq[q4].u;

  __shared__ float red[256];
  for (int c = 0; c < CQ_; ++c) {
    red[tid] = knv[c];
    __syncthreads();
    for (int off = 128; off > 0; off >>= 1) {
      if (tid < off) red[tid] += red[tid + off];
      __syncthreads();
    }
    if (tid == 0) atomicAdd(&ksum[b * CQ_ + c], red[0]);
    __syncthreads();
  }
}

// Kernel 4: vsum[b][c] = sum_n V[b][c][n]
__global__ __launch_bounds__(256)
void k4_vsum(const __bf16* __restrict__ vH, float* __restrict__ vsum)
{
  const int c = blockIdx.x, b = blockIdx.y, tid = threadIdx.x;
  const __bf16* p = vH + ((size_t)(b * COUT_ + c)) * N_;
  float s = 0.f;
  for (int n = tid; n < N_; n += 256) s += (float)p[n];
  __shared__ float red[256];
  red[tid] = s;
  __syncthreads();
  for (int off = 128; off > 0; off >>= 1) {
    if (tid < off) red[tid] += red[tid + off];
    __syncthreads();
  }
  if (tid == 0) vsum[b * COUT_ + c] = red[0];
}

// Kernel 5: tailor[b][n] = 1 / (N + sum_cq qn[b][n][cq] * (ksum[b][cq] + eps))
__global__ __launch_bounds__(256)
void k5_tailor(const __bf16* __restrict__ qnH, const float* __restrict__ ksum,
               float* __restrict__ tailor)
{
  const int b = blockIdx.y;
  const int n = blockIdx.x * 256 + threadIdx.x;
  const __bf16* q = qnH + ((size_t)(b * N_ + n)) * CQ_;
  float t = 0.f;
#pragma unroll
  for (int c = 0; c < CQ_; ++c) t += (float)q[c] * (ksum[b * CQ_ + c] + EPS_ATT);
  tailor[b * N_ + n] = 1.0f / ((float)N_ + t);
}

// =====================================================================
// Kernel 6: matrix[b][32][256] = Kn[32,N] x V^T[N,256], K-reduction over N.
// Direct global->fragment loads (no LDS). Output stored transposed bf16:
// matT[b][c][cq]. grid (4, B): 64 cols per block; 8 waves = 2(m) x 4(c).
// =====================================================================
__global__ __launch_bounds__(256)
void k6_matrix(const __bf16* __restrict__ knH, const __bf16* __restrict__ vH,
               __bf16* __restrict__ matT)
{
  const int b    = blockIdx.y;
  const int lane = threadIdx.x & 31;
  const int wave = threadIdx.x >> 5;
  const int mt   = (wave & 1) * 16;                    // cq tile
  const int c0   = blockIdx.x * 64 + (wave >> 1) * 16; // col tile

  const int rowA = mt + (lane & 15);
  const int kA   = (lane < 16) ? 0 : 8;
  const int colB = c0 + (lane & 15);
  const int kB   = (lane < 16) ? 0 : 8;

  const __bf16* aBase = knH + ((size_t)(b * CQ_ + rowA)) * N_;
  const __bf16* bBase = vH  + ((size_t)(b * COUT_ + colB)) * N_;

  f32x8 acc = (f32x8){0,0,0,0,0,0,0,0};
  for (int n0 = 0; n0 < N_; n0 += 32) {
    __builtin_prefetch(aBase + n0 + 512, 0, 0);
    __builtin_prefetch(bBase + n0 + 512, 0, 0);
    Frag a, bf;
    a.u[0]  = *(const uint4*)(aBase + n0 + kA);
    a.u[1]  = *(const uint4*)(aBase + n0 + kA + 16);
    bf.u[0] = *(const uint4*)(bBase + n0 + kB);
    bf.u[1] = *(const uint4*)(bBase + n0 + kB + 16);
    acc = wmma_bf16(a, bf, acc);
  }

  const int cn   = lane & 15;
  const int roff = (lane < 16) ? 0 : 8;
#pragma unroll
  for (int r = 0; r < 8; ++r) {
    int cq = mt + roff + r;
    int c  = c0 + cn;
    matT[((size_t)(b * COUT_ + c)) * CQ_ + cq] = (__bf16)acc[r];
  }
}

// =====================================================================
// Kernel 7: final fused stage.
// D[c,n] 16x16 = matT rows (A: 16c x 32cq) x qnH cols (B: 32cq x 16n),
// single 16x16x32 WMMA (K = Cq = 32 exactly). Epilogue:
//  ms = vsum[c] + D; wv = ms * tailor[n]; f = featF; out = f*(f + g*wv) + f.
// grid (N/64, B): 4 n-tiles per block; 8 waves x 2 c-tiles = 16 c-tiles.
// =====================================================================
__global__ __launch_bounds__(256)
void k7_final(const __bf16* __restrict__ matT, const __bf16* __restrict__ qnH,
              const float* __restrict__ featF, const float* __restrict__ vsum,
              const float* __restrict__ tailor, const float* __restrict__ gamma,
              float* __restrict__ out)
{
  const int b    = blockIdx.y;
  const int n0   = blockIdx.x * 64;
  const int lane = threadIdx.x & 31;
  const int wave = threadIdx.x >> 5;
  const float g  = gamma[0];

  const int kA = (lane < 16) ? 0 : 8;
  const int kB = (lane < 16) ? 0 : 8;
  const int cn = lane & 15;
  const int roff = (lane < 16) ? 0 : 8;

#pragma unroll
  for (int i = 0; i < 2; ++i) {
    const int cbase = (wave * 2 + i) * 16;
    Frag a;
    const __bf16* ap = matT + ((size_t)(b * COUT_ + cbase + cn)) * CQ_;
    a.u[0] = *(const uint4*)(ap + kA);
    a.u[1] = *(const uint4*)(ap + kA + 16);
#pragma unroll
    for (int j = 0; j < 4; ++j) {
      const int nt = n0 + j * 16;
      Frag bf;
      const __bf16* bp = qnH + ((size_t)(b * N_ + nt + cn)) * CQ_;
      bf.u[0] = *(const uint4*)(bp + kB);
      bf.u[1] = *(const uint4*)(bp + kB + 16);
      f32x8 acc = (f32x8){0,0,0,0,0,0,0,0};
      acc = wmma_bf16(a, bf, acc);

      const int nidx = nt + cn;             // D col = n (coalesced across lanes)
      const float tl = tailor[b * N_ + nidx];
#pragma unroll
      for (int r = 0; r < 8; ++r) {
        int c = cbase + roff + r;           // D row = c
        float ms = vsum[b * COUT_ + c] + acc[r];
        float wv = ms * tl;
        size_t idx = ((size_t)(b * COUT_ + c)) * N_ + nidx;
        float f = featF[idx];
        float att = f + g * wv;
        out[idx] = f * att + f;
      }
    }
  }
}

// Kernel 0: zero ksum
__global__ void k0_zero(float* p, int n) {
  int i = blockIdx.x * blockDim.x + threadIdx.x;
  if (i < n) p[i] = 0.0f;
}

// =====================================================================
extern "C" void kernel_launch(void* const* d_in, const int* in_sizes, int n_in,
                              void* d_out, int out_size, void* d_ws, size_t ws_size,
                              hipStream_t stream)
{
  const float* fsp   = (const float*)d_in[0];
  const float* fcp   = (const float*)d_in[1];
  const float* convw = (const float*)d_in[2];
  const float* bns   = (const float*)d_in[3];
  const float* bnb   = (const float*)d_in[4];
  const float* bnm   = (const float*)d_in[5];
  const float* bnv   = (const float*)d_in[6];
  const float* gamma = (const float*)d_in[7];
  const float* qw    = (const float*)d_in[8];
  const float* qb    = (const float*)d_in[9];
  const float* kw    = (const float*)d_in[10];
  const float* kb    = (const float*)d_in[11];
  const float* vw    = (const float*)d_in[12];
  const float* vb    = (const float*)d_in[13];
  float* out = (float*)d_out;

  char* ws = (char*)d_ws;
  const size_t szF  = (size_t)B_ * COUT_ * N_ * sizeof(float);   // 151 MB
  const size_t szH  = (size_t)B_ * COUT_ * N_ * sizeof(__bf16);  // 75.5 MB
  const size_t szQF = (size_t)B_ * CQ_ * N_ * sizeof(float);
  const size_t szQH = (size_t)B_ * N_ * CQ_ * sizeof(__bf16);

  float*  featF  = (float*) (ws);                 size_t off = szF;
  __bf16* featT  = (__bf16*)(ws + off);           off += szH;   // [b][n][c]
  __bf16* vH     = (__bf16*)(ws + off);           off += szH;
  float*  qF     = (float*) (ws + off);           off += szQF;
  float*  kF     = (float*) (ws + off);           off += szQF;
  __bf16* qnH    = (__bf16*)(ws + off);           off += szQH;
  __bf16* knH    = (__bf16*)(ws + off);           off += szQH;
  float*  ksum   = (float*) (ws + off);           off += (size_t)B_ * CQ_ * sizeof(float);
  float*  vsum   = (float*) (ws + off);           off += (size_t)B_ * COUT_ * sizeof(float);
  float*  tailor = (float*) (ws + off);           off += (size_t)B_ * N_ * sizeof(float);
  __bf16* matT   = (__bf16*)(ws + off);

  k0_zero<<<dim3(2), dim3(256), 0, stream>>>(ksum, B_ * CQ_);

  k1_conv_bn_relu<<<dim3(N_ / 128, COUT_ / 128, B_), dim3(256), 0, stream>>>(
      fsp, fcp, convw, bns, bnb, bnm, bnv, featF, featT);

  k2_qkv<<<dim3(N_ / 128, 3, B_), dim3(256), 0, stream>>>(
      featT, qw, qb, kw, kb, vw, vb, qF, kF, vH);

  k3_norm_qk<<<dim3(N_ / 256, B_), dim3(256), 0, stream>>>(qF, kF, qnH, knH, ksum);

  k4_vsum<<<dim3(COUT_, B_), dim3(256), 0, stream>>>(vH, vsum);

  k5_tailor<<<dim3(N_ / 256, B_), dim3(256), 0, stream>>>(qnH, ksum, tailor);

  k6_matrix<<<dim3(4, B_), dim3(256), 0, stream>>>(knH, vH, matT);

  k7_final<<<dim3(N_ / 64, B_), dim3(256), 0, stream>>>(
      matT, qnH, featF, vsum, tailor, gamma, out);
}
